// WideResNet_HyperDecisioNet_many_split_61349312856174
// MI455X (gfx1250) — matrix operand order repaired and checked by reading, not compiled, not run on hardware
//
#include <hip/hip_runtime.h>
#include <hip/hip_bf16.h>

// ---------------------------------------------------------------------------
// Types for CDNA5 WMMA / TDM
// ---------------------------------------------------------------------------
typedef __bf16 bf16_t;
typedef __attribute__((ext_vector_type(16))) __bf16 v16bf;
typedef __attribute__((ext_vector_type(8)))  float  v8f;
typedef unsigned int u32x4 __attribute__((ext_vector_type(4)));
typedef unsigned int u32x8 __attribute__((ext_vector_type(8)));

#define BATCH 64
#define EPSV 1e-5f

// ---------------------------------------------------------------------------
// gfx1250 async global->LDS copy (16B per lane), tracked by ASYNCcnt.
// ---------------------------------------------------------------------------
__device__ __forceinline__ void async_b128(unsigned lds_off, unsigned gl_off,
                                           const void* sbase)
{
    asm volatile("global_load_async_to_lds_b128 %0, %1, %2"
                 :: "v"(lds_off), "v"(gl_off), "s"(sbase) : "memory");
}
__device__ __forceinline__ void wait_async0()
{
    asm volatile("s_wait_asynccnt 0x0" ::: "memory");
}

// ---------------------------------------------------------------------------
// gfx1250 Tensor Data Mover: 2-D tile load, D# groups 0/1 in SGPRs.
// Tracked by TENSORcnt (wait via confirmed builtin).
// ---------------------------------------------------------------------------
__device__ __forceinline__ void tdm_load_2d(u32x4 g0, u32x8 g1)
{
    asm volatile("tensor_load_to_lds %0, %1" :: "s"(g0), "s"(g1) : "memory");
}

// ---------------------------------------------------------------------------
// Implicit-GEMM conv via v_wmma_f32_16x16x32_bf16.
//   D[Co, B*Ho*Wo] = Wbf16[128, Kp] * im2col(A)[K, B*Ho*Wo]
//   Block: 256 threads = 8 waves; tile 128(M) x 64(N); K chunked by 32.
//   A tile staged by TDM (3x3) or async-to-LDS (1x1) into double-buffered LDS.
//   Epilogue: f32 (+residual) OR fused BN+ReLU -> bf16.
// ---------------------------------------------------------------------------
template<int KS, int STRIDE, int PAD>
__global__ __launch_bounds__(256)
void k_conv_wmma(const bf16_t* __restrict__ act, const bf16_t* __restrict__ wgt,
                 float* __restrict__ out, const float* __restrict__ resid,
                 bf16_t* __restrict__ obf,
                 const float* __restrict__ bnG, const float* __restrict__ bnB,
                 const float* __restrict__ bnM, const float* __restrict__ bnV,
                 int Bn, int Ci, int H, int W, int Co, int Ho, int Wo, int Kp)
{
    __shared__ __align__(16) bf16_t As[2][128 * 32];   // 2 x 8 KB
    __shared__ __align__(16) bf16_t Bs[2][64  * 32];   // 2 x 4 KB

    const int tid  = threadIdx.x;
    const int wave = tid >> 5;
    const int lane = tid & 31;
    const int K    = Ci * KS * KS;          // logical K (<= Kp)
    const int Ntot = Bn * Ho * Wo;
    const int nBase = blockIdx.x * 64;
    const int mBase = blockIdx.y * 128 + wave * 16;

    // im2col coordinates this thread stages for the B tile (2 n's, 4 k's each)
    int  bI[2], oyI[2], oxI[2]; bool nokI[2];
    #pragma unroll
    for (int h = 0; h < 2; ++h) {
        const int nB = nBase + (tid >> 3) + h * 32;
        nokI[h] = nB < Ntot;
        int b = 0, oy = 0, ox = 0;
        if (nokI[h]) {
            int r = nB;
            b = r / (Ho * Wo); r -= b * (Ho * Wo);
            oy = r / Wo; ox = r - oy * Wo;
        }
        bI[h] = b; oyI[h] = oy * STRIDE - PAD; oxI[h] = ox * STRIDE - PAD;
    }

    auto stageA = [&](int buf, int k0) {
        if constexpr (KS == 1) {
            // small K: per-lane async copies (ASYNCcnt)
            #pragma unroll
            for (int r = 0; r < 2; ++r) {
                const int e   = tid + r * 256;
                const int row = e >> 2;
                const int c8  = (e & 3) * 8;
                unsigned loff = (unsigned)(size_t)(&As[buf][row * 32 + c8]);
                unsigned goff = (unsigned)((row * Kp + k0 + c8) * 2);
                async_b128(loff, goff, wgt);
            }
        } else {
            // big K: one TDM descriptor per chunk, issued by wave 0 (TENSORcnt)
            if (wave == 0) {
                const unsigned long long ga =
                    (unsigned long long)(size_t)wgt + (unsigned long long)(k0 * 2);
                u32x4 g0; u32x8 g1;
                g0[0] = 1u;                                     // count=1
                g0[1] = (unsigned)(size_t)(&As[buf][0]);        // lds_addr
                g0[2] = (unsigned)(ga & 0xFFFFFFFFu);           // global_addr lo
                g0[3] = ((unsigned)(ga >> 32) & 0x01FFFFFFu)
                        | 0x80000000u;                          // addr hi | type=2
                g1[0] = (1u << 16);                             // data_size = 2B
                g1[1] = ((unsigned)Kp) << 16;                   // tensor_dim0 lo16
                g1[2] = (((unsigned)Kp) >> 16) | (128u << 16);  // dim0 hi | dim1 lo
                g1[3] = (32u << 16);                            // dim1 hi | tile_dim0=32
                g1[4] = 128u;                                   // tile_dim1=128, tile_dim2=0
                g1[5] = (unsigned)Kp;                           // tensor_dim0_stride lo32
                g1[6] = 0u;
                g1[7] = 0u;
                tdm_load_2d(g0, g1);
            }
        }
    };
    auto waitA = [&]() {
        if constexpr (KS == 1) { wait_async0(); }
        else { if (wave == 0) __builtin_amdgcn_s_wait_tensorcnt(0); }
    };
    auto stageB = [&](int buf, int k0) {
        #pragma unroll
        for (int h = 0; h < 2; ++h) {
            const int nl = (tid >> 3) + h * 32;
            #pragma unroll
            for (int e = 0; e < 4; ++e) {
                const int kl = (tid & 7) * 4 + e;
                const int kg = k0 + kl;
                bf16_t v = (bf16_t)0.f;
                if (nokI[h] && kg < K) {
                    const int ci = kg / (KS * KS);
                    const int rr = kg - ci * (KS * KS);
                    const int ky = rr / KS, kx = rr - ky * KS;
                    const int iy = oyI[h] + ky, ix = oxI[h] + kx;
                    if (iy >= 0 && iy < H && ix >= 0 && ix < W)
                        v = act[(((size_t)bI[h] * Ci + ci) * H + iy) * W + ix];
                }
                Bs[buf][nl * 32 + kl] = v;
            }
        }
    };

    v8f zero = {0.f,0.f,0.f,0.f,0.f,0.f,0.f,0.f};
    v8f acc[4]; acc[0] = zero; acc[1] = zero; acc[2] = zero; acc[3] = zero;

    const int nchunks = Kp >> 5;
    stageA(0, 0); stageB(0, 0);
    waitA();
    __syncthreads();

    for (int c = 0; c < nchunks; ++c) {
        const int cur = c & 1;
        if (c + 1 < nchunks) { stageA(cur ^ 1, (c + 1) * 32); stageB(cur ^ 1, (c + 1) * 32); }

        // fragments per the 16-bit WMMA VGPR layouts (ISA 7.12.2)
        union { v16bf v; uint4 q[2]; } aF, bF;
        const int m   = lane & 15;
        const int klo = (lane < 16) ? 0 : 8;       // A: lanes<16 hold K0..7/16..23
        const bf16_t* ap = &As[cur][(wave * 16 + m) * 32];
        aF.q[0] = *(const uint4*)(ap + klo);
        aF.q[1] = *(const uint4*)(ap + klo + 16);
        const int kb = (lane < 16) ? 0 : 8;        // B: lanes<16 hold K0..7/16..23
        #pragma unroll
        for (int sub = 0; sub < 4; ++sub) {
            const bf16_t* bp = &Bs[cur][(sub * 16 + (lane & 15)) * 32 + kb];
            bF.q[0] = *(const uint4*)(bp);
            bF.q[1] = *(const uint4*)(bp + 16);
            acc[sub] = __builtin_amdgcn_wmma_f32_16x16x32_bf16(
                           false, aF.v, false, bF.v, (short)0, acc[sub], false, false);
        }

        if (c + 1 < nchunks) waitA();
        __syncthreads();
    }

    // epilogue: C/D layout — VGPR j: lanes0-15 M=j, lanes16-31 M=j+8
    #pragma unroll
    for (int sub = 0; sub < 4; ++sub) {
        const int n = nBase + sub * 16 + (lane & 15);
        if (n < Ntot) {
            int r = n;
            const int b2 = r / (Ho * Wo); r -= b2 * (Ho * Wo);
            const int oy2 = r / Wo, ox2 = r - oy2 * Wo;
            const int mo = (lane < 16) ? 0 : 8;
            #pragma unroll
            for (int j = 0; j < 8; ++j) {
                const int mm = mBase + mo + j;
                if (mm < Co) {
                    const size_t oi = (((size_t)b2 * Co + mm) * Ho + oy2) * Wo + ox2;
                    float vv = acc[sub][j];
                    if (obf) {                               // fused BN+ReLU -> bf16
                        float sc = bnG[mm] * rsqrtf(bnV[mm] + EPSV);
                        float t  = (vv - bnM[mm]) * sc + bnB[mm];
                        obf[oi] = (bf16_t)fmaxf(t, 0.f);
                    } else {
                        if (resid) vv += resid[oi];
                        out[oi] = vv;
                    }
                }
            }
        }
    }
}

// ---------------------------------------------------------------------------
// Weight prep: f32 (Co,K) -> bf16 (128, Kp) zero-padded panel
// ---------------------------------------------------------------------------
__global__ void k_wprep(const float* __restrict__ w, bf16_t* __restrict__ o,
                        int Co, int K, int Kp)
{
    int i = blockIdx.x * 256 + threadIdx.x;
    if (i >= 128 * Kp) return;
    int r = i / Kp, k = i - r * Kp;
    float v = (r < Co && k < K) ? w[(size_t)r * K + k] : 0.f;
    o[i] = (bf16_t)v;
}

// ---------------------------------------------------------------------------
// Elementwise helpers
// ---------------------------------------------------------------------------
__global__ void k_f32_to_bf16(const float* __restrict__ a, bf16_t* __restrict__ o, size_t n)
{
    size_t i = (size_t)blockIdx.x * 256 + threadIdx.x;
    if (i < n) o[i] = (bf16_t)a[i];
}

__global__ void k_bnrelu_bf16(const float* __restrict__ x, bf16_t* __restrict__ y,
                              const float* __restrict__ g, const float* __restrict__ b,
                              const float* __restrict__ m, const float* __restrict__ v,
                              int C, int HW, size_t n)
{
    size_t i = (size_t)blockIdx.x * 256 + threadIdx.x;
    if (i >= n) return;
    int c = (int)((i / (size_t)HW) % (size_t)C);
    float s = g[c] * rsqrtf(v[c] + EPSV);
    float t = (x[i] - m[c]) * s + b[c];
    y[i] = (bf16_t)fmaxf(t, 0.f);
}

// out = h_j(b) * cw + cb (+ sc)    with h = (sr>1e-4)*0.5 + sr/64
__global__ void k_hcombine(const float* __restrict__ cw, const float* __restrict__ cb,
                           const float* __restrict__ sc, const float* __restrict__ sr,
                           int j, float* __restrict__ out, int CHW, size_t n)
{
    size_t i = (size_t)blockIdx.x * 256 + threadIdx.x;
    if (i >= n) return;
    int b = (int)(i / (size_t)CHW);
    float s = sr[b * 2 + j];
    float h = (s > 1e-4f ? 0.5f : 0.f) + s * (1.f / 64.f);
    float val = h * cw[i] + cb[i];
    if (sc) val += sc[i];
    out[i] = val;
}

// fused: bf16( relu( bn( h_j*cw + cb ) ) )
__global__ void k_hcombine_bn(const float* __restrict__ cw, const float* __restrict__ cb,
                              const float* __restrict__ sr, int j,
                              const float* __restrict__ g, const float* __restrict__ b,
                              const float* __restrict__ m, const float* __restrict__ v,
                              bf16_t* __restrict__ out, int CHW, int HW, size_t n)
{
    size_t i = (size_t)blockIdx.x * 256 + threadIdx.x;
    if (i >= n) return;
    int bb = (int)(i / (size_t)CHW);
    int c  = (int)((i / (size_t)HW) % 128);
    float s = sr[bb * 2 + j];
    float h = (s > 1e-4f ? 0.5f : 0.f) + s * (1.f / 64.f);
    float val = h * cw[i] + cb[i];
    float sc  = g[c] * rsqrtf(v[c] + EPSV);
    float t   = (val - m[c]) * sc + b[c];
    out[i] = (bf16_t)fmaxf(t, 0.f);
}

__global__ void k_blend(const float* __restrict__ y0, const float* __restrict__ y1,
                        const float* __restrict__ sb, float* __restrict__ out,
                        int CHW, size_t n)
{
    size_t i = (size_t)blockIdx.x * 256 + threadIdx.x;
    if (i >= n) return;
    int b = (int)(i / (size_t)CHW);
    float s = sb[b];
    out[i] = (1.f - s) * y0[i] + s * y1[i];
}

// ebsl: feat = mean(x, HW); softmax(feat @ w.T + b); writes sr/sb + outputs
__global__ void k_ebsl(const float* __restrict__ x, const float* __restrict__ w,
                       const float* __restrict__ bias,
                       float* __restrict__ sr, float* __restrict__ sb,
                       float* __restrict__ osb, float* __restrict__ osr,
                       float* __restrict__ osbr, int C, int HW, int hIdx)
{
    __shared__ float feat[128];
    const int b = blockIdx.x, c = threadIdx.x;
    const float* xp = x + ((size_t)b * C + c) * HW;
    float s = 0.f;
    for (int k = 0; k < HW; ++k) s += xp[k];
    feat[c] = s / (float)HW;
    __syncthreads();
    if (c == 0) {
        float l0 = bias[0], l1 = bias[1];
        for (int k = 0; k < C; ++k) { l0 += feat[k] * w[k]; l1 += feat[k] * w[C + k]; }
        float mx = fmaxf(l0, l1);
        float e0 = __expf(l0 - mx), e1 = __expf(l1 - mx);
        float inv = 1.f / (e0 + e1);
        float s0 = e0 * inv, s1 = e1 * inv;
        float sbv = (s1 > 0.5f) ? 1.f : 0.f;
        sr[b * 2 + 0] = s0; sr[b * 2 + 1] = s1; sb[b] = sbv;
        osb[b * 6 + hIdx]          = sbv;
        osr[b * 12 + 2 * hIdx]     = s0;
        osr[b * 12 + 2 * hIdx + 1] = s1;
        osbr[b * 6 + hIdx]         = sbv;
    }
}

// final: relu(bn(x)) (S=1 so pool is identity) -> logits = t @ fc_w.T + fc_b
__global__ void k_final(const float* __restrict__ x,
                        const float* __restrict__ g, const float* __restrict__ b,
                        const float* __restrict__ m, const float* __restrict__ v,
                        const float* __restrict__ fcw, const float* __restrict__ fcb,
                        float* __restrict__ logits, int C, int NC)
{
    __shared__ float t[128];
    const int bidx = blockIdx.x, c = threadIdx.x;
    float s = g[c] * rsqrtf(v[c] + EPSV);
    t[c] = fmaxf((x[(size_t)bidx * C + c] - m[c]) * s + b[c], 0.f);
    __syncthreads();
    for (int o = threadIdx.x; o < NC; o += blockDim.x) {
        float a = fcb[o];
        for (int k = 0; k < C; ++k) a += t[k] * fcw[o * C + k];
        logits[(size_t)bidx * NC + o] = a;
    }
}

// ---------------------------------------------------------------------------
// Host orchestration
// ---------------------------------------------------------------------------
extern "C" void kernel_launch(void* const* d_in, const int* in_sizes, int n_in,
                              void* d_out, int out_size, void* d_ws, size_t ws_size,
                              hipStream_t stream)
{
    (void)in_sizes; (void)n_in; (void)out_size; (void)ws_size;
    const int Bn = BATCH;

    // ---- input unpacking: x first, then jax-pytree (sorted-key) param leaves
    int idx = 0;
    auto nxt = [&]() { return (const float*)d_in[idx++]; };
    struct BN { const float *b, *g, *m, *v; };
    auto rdBN = [&]() { BN p; p.b = nxt(); p.g = nxt(); p.m = nxt(); p.v = nxt(); return p; };

    const float* x_in = nxt();
    BN bn_final = rdBN();
    const float* conv1_w = nxt();
    const float* fc_b = nxt();
    const float* fc_w = nxt();
    struct HypP { BN bb1, bb2; const float *bbc1, *bbc2, *ebb, *ebw;
                  BN hb1, hb2; const float *h1b, *h1w, *h2b, *h2w, *hsc; } hyp[6];
    for (int i = 0; i < 6; ++i) {
        hyp[i].bb1 = rdBN(); hyp[i].bb2 = rdBN();
        hyp[i].bbc1 = nxt(); hyp[i].bbc2 = nxt();
        hyp[i].ebb = nxt();  hyp[i].ebw = nxt();
        hyp[i].hb1 = rdBN(); hyp[i].hb2 = rdBN();
        hyp[i].h1b = nxt();  hyp[i].h1w = nxt();
        hyp[i].h2b = nxt();  hyp[i].h2w = nxt();
        hyp[i].hsc = nxt();
    }
    struct L1P { BN bn1, bn2; const float *c1, *c2, *sc; } l1[4];
    for (int i = 0; i < 4; ++i) {
        l1[i].bn1 = rdBN(); l1[i].bn2 = rdBN();
        l1[i].c1 = nxt(); l1[i].c2 = nxt();
        l1[i].sc = (i == 0) ? nxt() : nullptr;
    }

    // ---- workspace carving
    const size_t SLOT = (size_t)Bn * 128 * 64 * 64 * sizeof(float);   // 134 MB
    char* base = (char*)d_ws;
    float*  fA = (float*)(base + 0 * SLOT);
    float*  fB = (float*)(base + 1 * SLOT);
    float*  fC = (float*)(base + 2 * SLOT);
    float*  fD = (float*)(base + 3 * SLOT);
    bf16_t* hA = (bf16_t*)(base + 4 * SLOT);
    bf16_t* hB = (bf16_t*)(base + 4 * SLOT + SLOT / 2);
    bf16_t* hC = (bf16_t*)(base + 5 * SLOT);
    float*  smallBuf = (float*)(base + 5 * SLOT + SLOT / 2);
    float*  srBuf = smallBuf;
    float*  sbBuf = smallBuf + 256;
    bf16_t* wBuf  = (bf16_t*)(smallBuf + 1024);     // padded bf16 weight panel (288 KB max)

    float* outF       = (float*)d_out;
    float* out_logits = outF;
    float* out_sb     = outF + 6400;
    float* out_sr     = outF + 6784;
    float* out_sbr    = outF + 7552;

    auto ewg = [](size_t n) { return dim3((unsigned)((n + 255) / 256)); };
    auto bnrelu = [&](const float* src, bf16_t* dst, const BN& p, int C, int HW) {
        size_t n = (size_t)Bn * C * HW;
        k_bnrelu_bf16<<<ewg(n), 256, 0, stream>>>(src, dst, p.g, p.b, p.m, p.v, C, HW, n);
    };
    auto cvt = [&](const float* src, bf16_t* dst, size_t n) {
        k_f32_to_bf16<<<ewg(n), 256, 0, stream>>>(src, dst, n);
    };
    auto convFull = [&](int ks, int stride, const bf16_t* a, const float* w,
                        float* o, const float* r, bf16_t* obf, const BN* bn,
                        int Ci, int H, int Co) {
        int Ho = (stride == 2) ? H / 2 : H, Wo = Ho;
        int K  = Ci * ks * ks;
        int Kp = ((K + 31) / 32) * 32;
        k_wprep<<<ewg((size_t)128 * Kp), 256, 0, stream>>>(w, wBuf, Co, K, Kp);
        size_t Ntot = (size_t)Bn * Ho * Wo;
        dim3 g((unsigned)((Ntot + 63) / 64), (unsigned)((Co + 127) / 128));
        const float* g_ = bn ? bn->g : nullptr; const float* b_ = bn ? bn->b : nullptr;
        const float* m_ = bn ? bn->m : nullptr; const float* v_ = bn ? bn->v : nullptr;
        if (ks == 3 && stride == 1)
            k_conv_wmma<3, 1, 1><<<g, 256, 0, stream>>>(a, wBuf, o, r, obf, g_, b_, m_, v_, Bn, Ci, H, H, Co, Ho, Wo, Kp);
        else if (ks == 3)
            k_conv_wmma<3, 2, 1><<<g, 256, 0, stream>>>(a, wBuf, o, r, obf, g_, b_, m_, v_, Bn, Ci, H, H, Co, Ho, Wo, Kp);
        else if (stride == 1)
            k_conv_wmma<1, 1, 0><<<g, 256, 0, stream>>>(a, wBuf, o, r, obf, g_, b_, m_, v_, Bn, Ci, H, H, Co, Ho, Wo, Kp);
        else
            k_conv_wmma<1, 2, 0><<<g, 256, 0, stream>>>(a, wBuf, o, r, obf, g_, b_, m_, v_, Bn, Ci, H, H, Co, Ho, Wo, Kp);
    };
    auto convK = [&](int ks, int stride, const bf16_t* a, const float* w, float* o,
                     const float* r, int Ci, int H, int Co) {
        convFull(ks, stride, a, w, o, r, nullptr, nullptr, Ci, H, Co);
    };
    auto convB = [&](int ks, int stride, const bf16_t* a, const float* w, bf16_t* obf,
                     const BN& bn, int Ci, int H, int Co) {
        convFull(ks, stride, a, w, nullptr, nullptr, obf, &bn, Ci, H, Co);
    };

    // ==== conv1: (B,3,64,64) -> (B,16,64,64)
    cvt(x_in, hA, (size_t)Bn * 3 * 64 * 64);
    convK(3, 1, hA, conv1_w, fA, nullptr, 3, 64, 16);

    // ==== layer1 block 0 (16 -> 128, shortcut 1x1); conv1 fuses bn2+relu
    bnrelu(fA, hB, l1[0].bn1, 16, 4096);
    convB(3, 1, hB, l1[0].c1, hC, l1[0].bn2, 16, 64, 128);      // u in hC
    cvt(fA, hA, (size_t)Bn * 16 * 4096);                        // xb
    convK(1, 1, hA, l1[0].sc, fC, nullptr, 16, 64, 128);        // sc
    convK(3, 1, hC, l1[0].c2, fD, fC, 128, 64, 128);            // y = conv2 + sc
    float* xcur = fD;

    // ==== layer1 blocks 1..3 (identity residual); conv1 fuses bn2+relu
    float* all4[4] = {fA, fB, fC, fD};
    for (int k = 1; k < 4; ++k) {
        float* s2 = nullptr;
        for (int q = 0; q < 4; ++q) if (all4[q] != xcur) { s2 = all4[q]; break; }
        bnrelu(xcur, hA, l1[k].bn1, 128, 4096);
        convB(3, 1, hA, l1[k].c1, hB, l1[k].bn2, 128, 64, 128);
        convK(3, 1, hB, l1[k].c2, s2, xcur, 128, 64, 128);
        xcur = s2;
    }

    // ==== hyper stages: per-sample conv rewritten as shared convs
    //      conv(x, h*W + bias) == h*conv(x,W) + conv(x,bias)
    int S = 64;
    for (int i = 0; i < 6; ++i) {
        HypP& H = hyp[i];
        const int So = S / 2;
        const int HWs = S * S, HWo = So * So;
        const size_t fsN = (size_t)Bn * 128 * HWs;
        const size_t foN = (size_t)Bn * 128 * HWo;

        float* slots3[3]; int cc = 0;
        for (int q = 0; q < 4; ++q) if (all4[q] != xcur) slots3[cc++] = all4[q];
        const size_t per = (SLOT / sizeof(float)) / foN;     // sub-buffers per slot (>=4)
        auto sub = [&](int k2) { return slots3[k2 / (int)per] + (size_t)(k2 % (int)per) * foN; };

        k_ebsl<<<Bn, 128, 0, stream>>>(xcur, H.ebw, H.ebb, srBuf, sbBuf,
                                       out_sb, out_sr, out_sbr, 128, HWs, i);

        bnrelu(xcur, hB, H.hb1, 128, HWs);       // t  = relu(bn1(x))  (shared by branches)
        cvt(xcur, hA, fsN);                      // xb = bf16(x)       (for shortcut conv)

        float *cw1 = sub(0), *cb1 = sub(1), *scb = sub(2);
        float *T[2] = {sub(3), sub(4)}, *Y[2] = {sub(5), sub(6)};
        float *R = sub(7), *Q = sub(8);

        convK(3, 2, hB, H.h1w, cw1, nullptr, 128, S, 128);   // conv(t, h1_w), stride2
        convK(3, 2, hB, H.h1b, cb1, nullptr, 128, S, 128);   // conv(t, h1_b), stride2
        convK(1, 2, hA, H.hsc, scb, nullptr, 128, S, 128);   // shortcut 1x1 stride2

        for (int j = 0; j < 2; ++j) {
            // u = relu(bn2(h*cw1 + cb1)) fused -> bf16
            k_hcombine_bn<<<ewg(foN), 256, 0, stream>>>(cw1, cb1, srBuf, j,
                H.hb2.g, H.hb2.b, H.hb2.m, H.hb2.v, hC, 128 * HWo, HWo, foN);
            convK(3, 1, hC, H.h2w, R, nullptr, 128, So, 128);
            convK(3, 1, hC, H.h2b, Q, nullptr, 128, So, 128);
            k_hcombine<<<ewg(foN), 256, 0, stream>>>(R, Q, scb, srBuf, j, T[j], 128 * HWo, foN);
            // bb basic block on branch j (identity residual); conv1 fuses bn2+relu
            bnrelu(T[j], hC, H.bb1, 128, HWo);
            convB(3, 1, hC, H.bbc1, hA, H.bb2, 128, So, 128);
            convK(3, 1, hA, H.bbc2, Y[j], T[j], 128, So, 128);
        }
        k_blend<<<ewg(foN), 256, 0, stream>>>(Y[0], Y[1], sbBuf, xcur, 128 * HWo, foN);
        S = So;
    }

    // ==== final BN + (1x1 pool) + FC
    k_final<<<Bn, 128, 0, stream>>>(xcur, bn_final.g, bn_final.b, bn_final.m, bn_final.v,
                                    fc_w, fc_b, out_logits, 128, 100);
}
